// GAT_MLP_42872363549080
// MI455X (gfx1250) — compile-verified, matrix operand order checked by reading, not run on hardware
//
#include <hip/hip_runtime.h>
#include <hip/hip_bf16.h>
#include <math.h>

// ---------------------------------------------------------------------------
// Problem constants (reference: B,S,N,E,OUT = 16,256,10000,320000,24)
// ---------------------------------------------------------------------------
#define BB   16
#define SS   256
#define NN   10000
#define EE   320000
#define OUTD 24
#define H1   4
#define C1   64
#define H2   4
#define C2   128

typedef __attribute__((ext_vector_type(16))) _Float16 v16h;
typedef __attribute__((ext_vector_type(8)))  float    v8f;
typedef __attribute__((ext_vector_type(8)))  _Float16 h8;

// ---------------------------------------------------------------------------
// WMMA fragment load: 16-bit A/B fragment from row-major [16 rows][K] memory.
// Lane L holds row (L&15); lanes 0-15 take K-chunks {0..7,16..23},
// lanes 16-31 take {8..15,24..31}  (CDNA5 ISA 7.12.2 16-bit A layout).
// Two global_load_b128 per lane.
// ---------------------------------------------------------------------------
__device__ __forceinline__ v16h load_frag16(const _Float16* __restrict__ base,
                                            int ld, int lane) {
  int row = lane & 15;
  int kh  = (lane >> 4) << 3;                 // 0 or 8
  const _Float16* p = base + (size_t)row * ld + kh;
  h8 lo = *(const h8*)p;                      // K = kh .. kh+7
  h8 hi = *(const h8*)(p + 16);               // K = kh+16 .. kh+23
  v16h f;
#pragma unroll
  for (int i = 0; i < 8; ++i) { f[i] = lo[i]; f[i + 8] = hi[i]; }
  return f;
}

// ---------------------------------------------------------------------------
// C = A(MxK,f16,row-major) * Bt(NcxK,f16,row-major == B^T) -> C(MxNc,f32)
// Each wave computes a 16x64 strip (4 accumulators) so one A-fragment feeds
// 4 WMMAs. Block = 8 waves = 2 m-tiles x 4 n-strips = 32x256 output tile.
// K multiple of 32; Nc multiple of 64; M multiple of 16.
// ---------------------------------------------------------------------------
__global__ void __launch_bounds__(256)
gemm_f16_wmma(const _Float16* __restrict__ A, const _Float16* __restrict__ Bt,
              float* __restrict__ C, int M, int K, int Nc) {
  int lane = threadIdx.x & 31;
  int w    = threadIdx.x >> 5;                 // 0..7
  int mi   = w >> 2;                           // 0..1
  int ni   = w & 3;                            // 0..3
  int mt   = blockIdx.x * 2 + mi;              // 16-row tile index
  int n64  = blockIdx.y * 4 + ni;              // 64-col strip index
  if (mt * 16 >= M || n64 * 64 >= Nc) return;  // whole-wave uniform exit
  const _Float16* Ab = A + (size_t)mt * 16 * K;
  const _Float16* Bb = Bt + (size_t)n64 * 64 * K;
  v8f acc0 = {}, acc1 = {}, acc2 = {}, acc3 = {};
  for (int k = 0; k < K; k += 32) {
    // prefetch next A chunk (global_prefetch_b8)
    __builtin_prefetch(Ab + k + 64, 0, 1);
    v16h a  = load_frag16(Ab + k, K, lane);
    v16h b0 = load_frag16(Bb + k, K, lane);
    v16h b1 = load_frag16(Bb + (size_t)16 * K + k, K, lane);
    v16h b2 = load_frag16(Bb + (size_t)32 * K + k, K, lane);
    v16h b3 = load_frag16(Bb + (size_t)48 * K + k, K, lane);
    acc0 = __builtin_amdgcn_wmma_f32_16x16x32_f16(false, a, false, b0,
                                                  (short)0, acc0, false, false);
    acc1 = __builtin_amdgcn_wmma_f32_16x16x32_f16(false, a, false, b1,
                                                  (short)0, acc1, false, false);
    acc2 = __builtin_amdgcn_wmma_f32_16x16x32_f16(false, a, false, b2,
                                                  (short)0, acc2, false, false);
    acc3 = __builtin_amdgcn_wmma_f32_16x16x32_f16(false, a, false, b3,
                                                  (short)0, acc3, false, false);
  }
  // C/D layout: VGPR r -> row r + 8*(lane>>4); col = lane&15
  int row0 = (lane >> 4) << 3;
  int col  = lane & 15;
  float* Cb = C + ((size_t)mt * 16 + row0) * Nc + (size_t)n64 * 64 + col;
#pragma unroll
  for (int r = 0; r < 8; ++r) {
    Cb[(size_t)r * Nc +  0] = acc0[r];
    Cb[(size_t)r * Nc + 16] = acc1[r];
    Cb[(size_t)r * Nc + 32] = acc2[r];
    Cb[(size_t)r * Nc + 48] = acc3[r];
  }
}

// ---------------------------------------------------------------------------
// Tiled transpose + f32 -> f16 cast:  in[z][rows][cols] -> out[z][cols][rows]
// ---------------------------------------------------------------------------
__global__ void __launch_bounds__(256)
transpose_cast_f16(const float* __restrict__ in, _Float16* __restrict__ out,
                   int rows, int cols) {
  __shared__ float tile[32][33];
  size_t slice = (size_t)rows * cols * blockIdx.z;
  const float* inz = in + slice;
  _Float16* outz = out + slice;
  int r0 = blockIdx.y * 32, c0 = blockIdx.x * 32;
  int tx = threadIdx.x, ty = threadIdx.y;     // (32,8)
#pragma unroll
  for (int i = 0; i < 4; ++i) {
    int r = r0 + ty + 8 * i, c = c0 + tx;
    if (r < rows && c < cols) tile[ty + 8 * i][tx] = inz[(size_t)r * cols + c];
  }
  __syncthreads();
#pragma unroll
  for (int i = 0; i < 4; ++i) {
    int c = c0 + ty + 8 * i, r = r0 + tx;
    if (c < cols && r < rows)
      outz[(size_t)c * rows + r] = (_Float16)tile[tx][ty + 8 * i];
  }
}

// ---------------------------------------------------------------------------
// CSR build (dst-indexed), rebuilt every call (in-row edge order set by atomics)
// ---------------------------------------------------------------------------
__global__ void zero_i32(int* p, int n) {
  int i = blockIdx.x * blockDim.x + threadIdx.x;
  if (i < n) p[i] = 0;
}
__global__ void deg_count(const int* __restrict__ ei, int E, int* deg) {
  int e = blockIdx.x * blockDim.x + threadIdx.x;
  if (e < E) atomicAdd(&deg[ei[E + e]], 1);   // dst row = ei[E..2E-1]
}
__global__ void deg_self(int* deg, int n) {
  int i = blockIdx.x * blockDim.x + threadIdx.x;
  if (i < n) deg[i] += 1;                     // self loop per node
}
// single-block exclusive scan: off[i] = sum(deg[0..i-1]), off[N] = total
__global__ void __launch_bounds__(1024)
scan_offsets(const int* __restrict__ deg, int* __restrict__ off, int n) {
  __shared__ int part[1024];
  int t = threadIdx.x;
  int chunk = (n + 1024) / 1024 + 1;          // covers index n inclusive
  int base = t * chunk;
  int s = 0;
  for (int i = 0; i < chunk; ++i) {
    int idx = base + i;
    if (idx < n) s += deg[idx];
  }
  part[t] = s;
  __syncthreads();
  if (t == 0) {
    int run = 0;
    for (int i = 0; i < 1024; ++i) { int v = part[i]; part[i] = run; run += v; }
  }
  __syncthreads();
  int run = part[t];
  for (int i = 0; i < chunk; ++i) {
    int idx = base + i;
    if (idx < n) { off[idx] = run; run += deg[idx]; }
    else if (idx == n) off[idx] = run;
  }
}
__global__ void scatter_edges(const int* __restrict__ ei, int E,
                              const int* __restrict__ off, int* cursor,
                              int* __restrict__ csr_src) {
  int e = blockIdx.x * blockDim.x + threadIdx.x;
  if (e < E) {
    int s = ei[e], d = ei[E + e];
    int p = atomicAdd(&cursor[d], 1);
    csr_src[off[d] + p] = s;
  }
}
__global__ void scatter_loops(const int* __restrict__ off, int* cursor,
                              int* __restrict__ csr_src, int n) {
  int i = blockIdx.x * blockDim.x + threadIdx.x;
  if (i < n) {
    int p = atomicAdd(&cursor[i], 1);
    csr_src[off[i] + p] = i;
  }
}

// ---------------------------------------------------------------------------
// Attention logit partial sums: ssum[n,h] = <h[n,h,:], as[h,:]>, same for dsum
// ---------------------------------------------------------------------------
__global__ void att_logit_sums(const float* __restrict__ h,
                               const float* __restrict__ as,
                               const float* __restrict__ ad,
                               float* __restrict__ ssum,
                               float* __restrict__ dsum, int n, int H, int C) {
  int i = blockIdx.x * blockDim.x + threadIdx.x;
  if (i >= n * H) return;
  int node = i / H, hh = i % H;
  const float* hp = h + (size_t)node * H * C + (size_t)hh * C;
  const float* ap = as + hh * C;
  const float* dp = ad + hh * C;
  float sa = 0.f, sd = 0.f;
  for (int c = 0; c < C; ++c) { float v = hp[c]; sa += v * ap[c]; sd += v * dp[c]; }
  ssum[i] = sa;
  dsum[i] = sd;
}

// ---------------------------------------------------------------------------
// GAT softmax-aggregate: one wave per destination node, loop over heads.
// out = mean_h softmax_dst(leakyrelu(ssum[src]+dsum[dst])) * h[src]  + bias
// Two passes per head: max, then exp-weighted sum (alpha applied via 1/sum).
// ---------------------------------------------------------------------------
template <int C, int H, bool ELU>
__global__ void __launch_bounds__(256)
gat_aggregate(const float* __restrict__ h, const float* __restrict__ ssum,
              const float* __restrict__ dsum, const int* __restrict__ off,
              const int* __restrict__ src, const float* __restrict__ bias,
              _Float16* __restrict__ out16, int n) {
  int node = blockIdx.x * 8 + (threadIdx.x >> 5);
  if (node >= n) return;                      // whole-wave uniform exit
  int lane = threadIdx.x & 31;
  constexpr int CPL = C / 32;
  float outc[CPL];
#pragma unroll
  for (int q = 0; q < CPL; ++q) outc[q] = 0.f;
  int e0 = off[node], e1 = off[node + 1];
#pragma unroll
  for (int hh = 0; hh < H; ++hh) {
    float d = dsum[node * H + hh];
    float m = -1e30f;
    for (int e = e0; e < e1; ++e) {
      float l = ssum[src[e] * H + hh] + d;
      l = (l > 0.f) ? l : 0.2f * l;           // LeakyReLU(0.2)
      m = fmaxf(m, l);
    }
    float acc[CPL];
#pragma unroll
    for (int q = 0; q < CPL; ++q) acc[q] = 0.f;
    float se = 0.f;
    for (int e = e0; e < e1; ++e) {
      int s = src[e];
      float l = ssum[s * H + hh] + d;
      l = (l > 0.f) ? l : 0.2f * l;
      float ex = __expf(l - m);
      se += ex;
      const float* hp = h + (size_t)s * H * C + (size_t)hh * C + lane;
#pragma unroll
      for (int q = 0; q < CPL; ++q) acc[q] += ex * hp[q * 32];
    }
    float inv = 1.f / (se + 1e-16f);
#pragma unroll
    for (int q = 0; q < CPL; ++q) outc[q] += acc[q] * inv;
  }
#pragma unroll
  for (int q = 0; q < CPL; ++q) {
    int c = lane + q * 32;
    float v = outc[q] * (1.f / H) + bias[c];
    if (ELU) v = (v > 0.f) ? v : (__expf(v) - 1.f);
    out16[(size_t)node * C + c] = (_Float16)v;
  }
}

// ---------------------------------------------------------------------------
// Per-node MLP: pred[n,b,:] = relu(out2[b,n,:] @ fw1[n] + fb1[n]) @ fw2[n]+fb2
// One block per node; memory-bound on the single fw1 stream (327 MB).
// ---------------------------------------------------------------------------
__global__ void __launch_bounds__(256)
node_mlp(const _Float16* __restrict__ out2 /*[B][N][C2] f16*/,
         const float* __restrict__ fw1, const float* __restrict__ fb1,
         const float* __restrict__ fw2, const float* __restrict__ fb2,
         float* __restrict__ pred /*[N][B][OUT]*/) {
  __shared__ float A[BB][C2];
  __shared__ float Y[BB][64];
  int n = blockIdx.x;
  for (int i = threadIdx.x; i < BB * C2; i += 256) {
    int b = i >> 7, f = i & (C2 - 1);
    A[b][f] = (float)out2[((size_t)b * NN + n) * C2 + f];
  }
  __syncthreads();
  const float* w1 = fw1 + (size_t)n * C2 * 64;
  for (int i = threadIdx.x; i < BB * 64; i += 256) {
    int b = i >> 6, o = i & 63;
    float s = fb1[(size_t)n * 64 + o];
    for (int k = 0; k < C2; ++k) s += A[b][k] * w1[(size_t)k * 64 + o];
    Y[b][o] = fmaxf(s, 0.f);
  }
  __syncthreads();
  const float* w2 = fw2 + (size_t)n * 64 * OUTD;
  for (int i = threadIdx.x; i < BB * OUTD; i += 256) {
    int b = i / OUTD, o = i % OUTD;
    float s = fb2[(size_t)n * OUTD + o];
    for (int k = 0; k < 64; ++k) s += Y[b][k] * w2[(size_t)k * OUTD + o];
    pred[((size_t)n * BB + b) * OUTD + o] = s;
  }
}

// ---------------------------------------------------------------------------
// Host orchestration
// ---------------------------------------------------------------------------
static inline size_t alignup(size_t x) { return (x + 255) & ~(size_t)255; }
static inline int ceildiv(int a, int b) { return (a + b - 1) / b; }

extern "C" void kernel_launch(void* const* d_in, const int* in_sizes, int n_in,
                              void* d_out, int out_size, void* d_ws,
                              size_t ws_size, hipStream_t stream) {
  const float* x   = (const float*)d_in[0];   // [B][S][N]
  const int*   ei  = (const int*)d_in[1];     // [2][E]
  const float* W1  = (const float*)d_in[2];   // [S][H1*C1]
  const float* as1 = (const float*)d_in[3];
  const float* ad1 = (const float*)d_in[4];
  const float* b1  = (const float*)d_in[5];
  const float* W2  = (const float*)d_in[6];   // [C1][H2*C2]
  const float* as2 = (const float*)d_in[7];
  const float* ad2 = (const float*)d_in[8];
  const float* b2  = (const float*)d_in[9];
  const float* fw1 = (const float*)d_in[10];
  const float* fb1 = (const float*)d_in[11];
  const float* fw2 = (const float*)d_in[12];
  const float* fb2 = (const float*)d_in[13];
  float* pred = (float*)d_out;

  const int E = in_sizes[1] / 2;              // 320000
  const int NE = E + NN;                      // edges incl. self loops
  const int F1 = H1 * C1;                     // 256
  const int F2 = H2 * C2;                     // 512

  // ---- workspace carve-up ----
  char* w = (char*)d_ws;
  size_t o = 0;
  _Float16* xh   = (_Float16*)(w + o); o = alignup(o + (size_t)BB * NN * SS * 2);
  _Float16* W1t  = (_Float16*)(w + o); o = alignup(o + (size_t)F1 * SS * 2);
  _Float16* W2t  = (_Float16*)(w + o); o = alignup(o + (size_t)F2 * C1 * 2);
  float*    h1   = (float*)(w + o);    o = alignup(o + (size_t)NN * F1 * 4);
  float*    h2   = (float*)(w + o);    o = alignup(o + (size_t)NN * F2 * 4);
  _Float16* x2h  = (_Float16*)(w + o); o = alignup(o + (size_t)NN * C1 * 2);
  _Float16* o2a  = (_Float16*)(w + o); o = alignup(o + (size_t)BB * NN * C2 * 2);
  float*    ssum = (float*)(w + o);    o = alignup(o + (size_t)NN * 4 * 4);
  float*    dsum = (float*)(w + o);    o = alignup(o + (size_t)NN * 4 * 4);
  int*      deg  = (int*)(w + o);      o = alignup(o + (size_t)NN * 4);
  int*      off  = (int*)(w + o);      o = alignup(o + (size_t)(NN + 1) * 4);
  int*      cur  = (int*)(w + o);      o = alignup(o + (size_t)NN * 4);
  int*      csrc = (int*)(w + o);      o = alignup(o + (size_t)NE * 4);
  (void)ws_size; (void)n_in; (void)out_size;

  // ---- stage 0: transpose/cast inputs to f16 ----
  {
    dim3 blk(32, 8);
    dim3 g0(ceildiv(NN, 32), ceildiv(SS, 32), BB);
    transpose_cast_f16<<<g0, blk, 0, stream>>>(x, xh, SS, NN);      // x[b][s][n] -> xh[b][n][s]
    dim3 g1(ceildiv(F1, 32), ceildiv(SS, 32), 1);
    transpose_cast_f16<<<g1, blk, 0, stream>>>(W1, W1t, SS, F1);    // W1t[out][k]
    dim3 g2(ceildiv(F2, 32), ceildiv(C1, 32), 1);
    transpose_cast_f16<<<g2, blk, 0, stream>>>(W2, W2t, C1, F2);    // W2t[out][k]
  }

  // ---- stage 1: CSR (dst-indexed, with self loops) ----
  zero_i32<<<ceildiv(NN, 256), 256, 0, stream>>>(deg, NN);
  deg_count<<<ceildiv(E, 256), 256, 0, stream>>>(ei, E, deg);
  deg_self<<<ceildiv(NN, 256), 256, 0, stream>>>(deg, NN);
  scan_offsets<<<1, 1024, 0, stream>>>(deg, off, NN);
  zero_i32<<<ceildiv(NN, 256), 256, 0, stream>>>(cur, NN);
  scatter_edges<<<ceildiv(E, 256), 256, 0, stream>>>(ei, E, off, cur, csrc);
  scatter_loops<<<ceildiv(NN, 256), 256, 0, stream>>>(off, cur, csrc, NN);

  // ---- stage 2: per-batch GAT layers ----
  const int gx = ceildiv(NN, 32);             // 313 blocks of 32 rows
  for (int b = 0; b < BB; ++b) {
    const _Float16* xb = xh + (size_t)b * NN * SS;
    // layer 1 GEMM: [N x S] @ [S x 256] -> h1
    gemm_f16_wmma<<<dim3(gx, F1 / 256), 256, 0, stream>>>(xb, W1t, h1, NN, SS, F1);
    att_logit_sums<<<ceildiv(NN * H1, 256), 256, 0, stream>>>(h1, as1, ad1, ssum, dsum, NN, H1, C1);
    gat_aggregate<C1, H1, true><<<ceildiv(NN, 8), 256, 0, stream>>>(
        h1, ssum, dsum, off, csrc, b1, x2h, NN);
    // layer 2 GEMM: [N x 64] @ [64 x 512] -> h2
    gemm_f16_wmma<<<dim3(gx, F2 / 256), 256, 0, stream>>>(x2h, W2t, h2, NN, C1, F2);
    att_logit_sums<<<ceildiv(NN * H2, 256), 256, 0, stream>>>(h2, as2, ad2, ssum, dsum, NN, H2, C2);
    gat_aggregate<C2, H2, false><<<ceildiv(NN, 8), 256, 0, stream>>>(
        h2, ssum, dsum, off, csrc, b2, o2a + (size_t)b * NN * C2, NN);
  }

  // ---- stage 3: per-node MLP -> pred [N][B][OUT] ----
  node_mlp<<<NN, 256, 0, stream>>>(o2a, fw1, fb1, fw2, fb2, pred);
}
